// MMAttention_59554016526670
// MI455X (gfx1250) — matrix operand order repaired and checked by reading, not compiled, not run on hardware
//
#include <hip/hip_runtime.h>

// ---------------------------------------------------------------------------
// MMAttention for MI455X (gfx1250, wave32, WMMA bf16 + f32 accum, async->LDS)
//
// b=8, n=1315 (291 image + 1024 gene), dim=512, h=8, d=64, conv kernel=33.
// Reference semantics replicated exactly, including the key permutation
// [k_gene, k_img] for scores, the two-stage softmax for image queries, V in
// ORIGINAL order for A@V, and the mixed-order out+res add.
//
// Fragment-contiguity trick (16-bit WMMA operand layouts):
//   A-frag lane data = rows [khalf*8,+8) and [16+khalf*8,+8)  -> 2x b128 loads
//   B-frag lane data = rows [khalf*8,+8) and [16+khalf*8,+8)                    -> 2x b128 loads
// so all operands are stored so per-lane fragment data is contiguous.
// ---------------------------------------------------------------------------

#define B_      8
#define NSEQ    1315
#define DIM     512
#define HEADS_  8
#define DHEAD   64
#define NROWS   (B_ * NSEQ)          // 10520
#define NQKV    (3 * HEADS_ * DHEAD) // 1536
#define NIMG    291
#define NGENE   1024
#define QTILES  83                   // ceil(1315/16)
#define KTILES  84
#define NPAD    (KTILES * 16)        // 1344

typedef __attribute__((ext_vector_type(16))) __bf16 v16bf;
typedef __attribute__((ext_vector_type(8)))  __bf16 v8bf;
typedef __attribute__((ext_vector_type(8)))  float  v8f;
typedef __attribute__((ext_vector_type(4)))  int    v4i;

typedef __attribute__((address_space(1))) v4i* gptr_v4i;   // global
typedef __attribute__((address_space(3))) v4i* lptr_v4i;   // LDS

static __device__ inline v8f wmma_bf16(v16bf a, v16bf b, v8f c) {
  return __builtin_amdgcn_wmma_f32_16x16x32_bf16(
      false, a, false, b, (short)0, c, false, false);
}

static __device__ inline v16bf cat16(v8bf lo, v8bf hi) {
  return __builtin_shufflevector(lo, hi, 0, 1, 2, 3, 4, 5, 6, 7,
                                         8, 9, 10, 11, 12, 13, 14, 15);
}
// A-frag from a contiguous 32-element (bf16) row segment at `p`
static __device__ inline v16bf a_frag(const __bf16* p, int khalf) {
  v8bf lo = *(const v8bf*)(p + khalf * 8);
  v8bf hi = *(const v8bf*)(p + 16 + khalf * 8);
  return cat16(lo, hi);
}
// B-frag from a contiguous 32-element (bf16) column segment at `p`
static __device__ inline v16bf b_frag(const __bf16* p, int khalf) {
  v8bf lo = *(const v8bf*)(p + khalf * 8);
  v8bf hi = *(const v8bf*)(p + 16 + khalf * 8);
  return cat16(lo, hi);
}

static __device__ inline float redmax8(float v) {
  v = fmaxf(v, __shfl_xor(v, 1, 32));
  v = fmaxf(v, __shfl_xor(v, 2, 32));
  v = fmaxf(v, __shfl_xor(v, 4, 32));
  return v;
}
static __device__ inline float redsum8(float v) {
  v += __shfl_xor(v, 1, 32);
  v += __shfl_xor(v, 2, 32);
  v += __shfl_xor(v, 4, 32);
  return v;
}

// CDNA5 async global->LDS (16B per lane), with synchronous fallback.
static __device__ inline void async_copy16(const __bf16* g, __bf16* l) {
#if __has_builtin(__builtin_amdgcn_global_load_async_to_lds_b128)
  __builtin_amdgcn_global_load_async_to_lds_b128(
      (gptr_v4i)g, (lptr_v4i)l, 0, 0);
#else
  *(v8bf*)l = *(const v8bf*)g;
#endif
}
static __device__ inline void async_wait0() {
#if __has_builtin(__builtin_amdgcn_s_wait_asynccnt)
  __builtin_amdgcn_s_wait_asynccnt(0);
#endif
}

// ---------------------------------------------------------------------------
// Kernel 0: one-time conversions. xb = bf16(x) (linear), wt = bf16(w_qkv)^T
// so GEMM B-fragments read contiguous runs of wt[c][k].
// ---------------------------------------------------------------------------
__global__ void prep_kernel(const float* __restrict__ x,
                            const float* __restrict__ w,
                            __bf16* __restrict__ xb,
                            __bf16* __restrict__ wt) {
  int i = blockIdx.x * 256 + threadIdx.x;
  if (i < NROWS * DIM) xb[i] = (__bf16)x[i];
  if (i < DIM * NQKV) {
    int k = i / NQKV, c = i - k * NQKV;
    wt[(size_t)c * DIM + k] = (__bf16)w[i];
  }
}

// ---------------------------------------------------------------------------
// Kernel 1: qkv = x @ w_qkv via WMMA. Block = 128 thr (4 waves); block tile =
// 16 rows x 256 cols, wave tile = 16x64 (4 accumulators). The shared 16x512
// bf16 A-slab is staged to LDS with async global->LDS loads.
// Outputs: qb (scaled, bf16 linear), kb (bf16 linear),
//          vt (bf16 TRANSPOSED [b][h][d][NPAD]), vf (f32 linear, for conv).
// ---------------------------------------------------------------------------
__global__ void qkv_wmma_kernel(const __bf16* __restrict__ xb,
                                const __bf16* __restrict__ wt,
                                __bf16* __restrict__ qb,
                                __bf16* __restrict__ kb,
                                __bf16* __restrict__ vt,
                                float*  __restrict__ vf) {
  __shared__ __bf16 xtile[16 * DIM];            // 16 KB
  const int lane  = threadIdx.x & 31;
  const int wave  = threadIdx.x >> 5;
  const int mrow  = lane & 15;
  const int khalf = lane >> 4;
  const int rtile = blockIdx.x;
  const int col0  = blockIdx.y * 256 + wave * 64;
  (void)lane;

  // ---- async stage A-slab: 16 rows x 512 bf16 = 1024 x 16B chunks, 64/row
#pragma unroll
  for (int i = 0; i < 8; ++i) {
    int chunk = threadIdx.x + 128 * i;          // 0..1023
    int rr = chunk >> 6;                        // row in tile
    int cc = chunk & 63;                        // 16B unit in row
    int rg = rtile * 16 + rr;
    int rs = (rg < NROWS) ? rg : 0;
    async_copy16(xb + (size_t)rs * DIM + cc * 8, xtile + rr * DIM + cc * 8);
  }
  async_wait0();
  __syncthreads();

  v8f acc0 = {}, acc1 = {}, acc2 = {}, acc3 = {};
  const __bf16* w0 = wt + (size_t)(col0 + 0  + mrow) * DIM;
  const __bf16* w1 = wt + (size_t)(col0 + 16 + mrow) * DIM;
  const __bf16* w2 = wt + (size_t)(col0 + 32 + mrow) * DIM;
  const __bf16* w3 = wt + (size_t)(col0 + 48 + mrow) * DIM;
  const __bf16* arow = xtile + mrow * DIM;

  for (int kk = 0; kk < DIM; kk += 32) {
    v16bf a  = a_frag(arow + kk, khalf);
    acc0 = wmma_bf16(a, b_frag(w0 + kk, khalf), acc0);
    acc1 = wmma_bf16(a, b_frag(w1 + kk, khalf), acc1);
    acc2 = wmma_bf16(a, b_frag(w2 + kk, khalf), acc2);
    acc3 = wmma_bf16(a, b_frag(w3 + kk, khalf), acc3);
  }

  // ---- scatter: col0 is 64-aligned, so section/head are uniform per wave
  const int sec = col0 >> 9;                    // 0=q 1=k 2=v
  const int hh  = (col0 >> 6) & 7;
  v8f accs[4] = {acc0, acc1, acc2, acc3};
#pragma unroll
  for (int t = 0; t < 4; ++t) {
    const int dd = t * 16 + mrow;
#pragma unroll
    for (int rr = 0; rr < 8; ++rr) {
      int rowg = rtile * 16 + rr + 8 * khalf;   // C layout: M = rr + 8*khalf
      if (rowg < NROWS) {
        int bb = rowg / NSEQ;
        int nn = rowg - bb * NSEQ;
        int bh = bb * HEADS_ + hh;
        float val = accs[t][rr];
        if (sec == 0)
          qb[((size_t)bh * NSEQ + nn) * DHEAD + dd] = (__bf16)(val * 0.125f);
        else if (sec == 1)
          kb[((size_t)bh * NSEQ + nn) * DHEAD + dd] = (__bf16)val;
        else {
          vt[((size_t)bh * DHEAD + dd) * NPAD + nn] = (__bf16)val;
          vf[((size_t)bh * NSEQ + nn) * DHEAD + dd] = val;
        }
      }
    }
  }
}

// ---------------------------------------------------------------------------
// Kernel 2: fused attention. Workgroup (4 waves) per (b*h, 16-query tile).
// Phase 1: S = Q_perm K_perm^T via WMMA -> f32 LDS [16][1344]
// Phase 2: exact (two-stage for image queries) softmax -> bf16 LDS [16][1344]
// Phase 3: O = P V via WMMA; P from ds_load_b128, V^T from global b128.
// Dynamic LDS: 86016 + 43008 = 129024 B (< 320 KB WGP LDS).
// ---------------------------------------------------------------------------
__global__ void attn_wmma_kernel(const __bf16* __restrict__ qb,
                                 const __bf16* __restrict__ kb,
                                 const __bf16* __restrict__ vt,
                                 float* __restrict__ out) {
  extern __shared__ char smem_raw[];
  float*  sc = (float*)smem_raw;                         // [16][NPAD] f32
  __bf16* pb = (__bf16*)(smem_raw + 16 * NPAD * 4);      // [16][NPAD] bf16

  const int lane  = threadIdx.x & 31;
  const int wave  = threadIdx.x >> 5;
  const int mrow  = lane & 15;
  const int khalf = lane >> 4;
  const int qt = blockIdx.x;
  const int bh = blockIdx.y;
  const int hh = bh & 7;
  const int bb = bh >> 3;
  const size_t base = (size_t)bh * NSEQ * DHEAD;
  const int o0 = qt * 16;

  // Q fragments (concat-order row -> original index; boundary is tile-aligned)
  int oq = o0 + mrow;
  int qn = 0;
  if (oq < NSEQ) qn = (oq < NGENE) ? (oq + NIMG) : (oq - NGENE);
  const __bf16* qrow = qb + base + (size_t)qn * DHEAD;
  v16bf aq0 = a_frag(qrow, khalf);
  v16bf aq1 = a_frag(qrow + 32, khalf);

  // ---- Phase 1: scores
  for (int t = wave; t < KTILES; t += 4) {
    int j = t * 16 + mrow;                      // permuted key column
    int kn = 0;
    if (j < NSEQ) kn = (j < NGENE) ? (j + NIMG) : (j - NGENE);
    const __bf16* krow = kb + base + (size_t)kn * DHEAD;
    v8f s = {};
    s = wmma_bf16(aq0, b_frag(krow, khalf), s);
    s = wmma_bf16(aq1, b_frag(krow + 32, khalf), s);
#pragma unroll
    for (int rr = 0; rr < 8; ++rr)
      sc[(rr + 8 * khalf) * NPAD + t * 16 + mrow] = s[rr];
  }
  __syncthreads();

  // ---- Phase 2: softmax (8 threads/row; strided elems are thread-private)
  {
    const int row  = threadIdx.x >> 3;
    const int slot = threadIdx.x & 7;
    float*  srow = sc + row * NPAD;
    __bf16* prow = pb + row * NPAD;
    int orow = o0 + row;
    if (orow < NSEQ) {
      if (orow >= NGENE) {                      // image query: pre-softmax gene blk
        float m1 = -3.4e38f;
        for (int j = slot; j < NGENE; j += 8) m1 = fmaxf(m1, srow[j]);
        m1 = redmax8(m1);
        float s1 = 0.f;
        for (int j = slot; j < NGENE; j += 8) s1 += __expf(srow[j] - m1);
        s1 = redsum8(s1);
        float inv1 = 1.0f / s1;
        for (int j = slot; j < NGENE; j += 8) srow[j] = __expf(srow[j] - m1) * inv1;
      }
      float m2 = -3.4e38f;
      for (int j = slot; j < NSEQ; j += 8) m2 = fmaxf(m2, srow[j]);
      m2 = redmax8(m2);
      float s2 = 0.f;
      for (int j = slot; j < NSEQ; j += 8) s2 += __expf(srow[j] - m2);
      s2 = redsum8(s2);
      float inv2 = 1.0f / s2;
      for (int j = slot; j < NSEQ; j += 8)
        prow[j] = (__bf16)(__expf(srow[j] - m2) * inv2);
      for (int j = NSEQ + slot; j < NPAD; j += 8) prow[j] = (__bf16)0.0f;
    } else {
      for (int j = slot; j < NPAD; j += 8) prow[j] = (__bf16)0.0f;
    }
  }
  __syncthreads();

  // ---- Phase 3: O = P @ V (V in ORIGINAL order via transposed vt)
  const int dd0 = wave * 16;
  const __bf16* vrow = vt + ((size_t)bh * DHEAD + dd0 + mrow) * NPAD;
  const __bf16* prow = pb + mrow * NPAD;
  v8f acc = {};
  for (int kt = 0; kt < NPAD; kt += 32) {
    v16bf ap = a_frag(prow + kt, khalf);        // 2x ds_load_b128
    v16bf bv = b_frag(vrow + kt, khalf);        // 2x global b128
    acc = wmma_bf16(ap, bv, acc);
  }
#pragma unroll
  for (int rr = 0; rr < 8; ++rr) {
    int o = o0 + rr + 8 * khalf;
    if (o < NSEQ)
      out[((size_t)bb * NSEQ + o) * (HEADS_ * DHEAD) + hh * DHEAD + dd0 + mrow]
          = acc[rr];
  }
}

// ---------------------------------------------------------------------------
// Kernel 3: depthwise 33-tap conv of V along n (original order) added into out
// at the same row index (replicates reference's mixed-order add). float4/thr.
// ---------------------------------------------------------------------------
__global__ void conv_add_kernel(const float* __restrict__ vf,
                                const float* __restrict__ wres,
                                float* __restrict__ out) {
  int idx = blockIdx.x * blockDim.x + threadIdx.x;   // over nE/4
  if (idx >= (B_ * HEADS_ * NSEQ * DHEAD) / 4) return;
  int dd4  = (idx & 15) * 4;
  int rest = idx >> 4;
  int o    = rest % NSEQ;
  int bh   = rest / NSEQ;
  int hh   = bh & 7;
  int bb   = bh >> 3;
  const float* vrow = vf + (size_t)bh * NSEQ * DHEAD;
  float4 s = make_float4(0.f, 0.f, 0.f, 0.f);
#pragma unroll
  for (int t = 0; t < 33; ++t) {
    int r = o - 16 + t;
    if (r >= 0 && r < NSEQ) {
      float wv = wres[hh * 33 + t];
      float4 v = *(const float4*)(vrow + (size_t)r * DHEAD + dd4);
      s.x += wv * v.x; s.y += wv * v.y; s.z += wv * v.z; s.w += wv * v.w;
    }
  }
  float* po = out + ((size_t)bb * NSEQ + o) * (HEADS_ * DHEAD) + hh * DHEAD + dd4;
  float4 cur = *(const float4*)po;
  cur.x += s.x; cur.y += s.y; cur.z += s.z; cur.w += s.w;
  *(float4*)po = cur;
}

// ---------------------------------------------------------------------------
extern "C" void kernel_launch(void* const* d_in, const int* in_sizes, int n_in,
                              void* d_out, int out_size, void* d_ws, size_t ws_size,
                              hipStream_t stream) {
  (void)in_sizes; (void)n_in; (void)out_size; (void)ws_size;
  const float* x     = (const float*)d_in[0];
  const float* w_qkv = (const float*)d_in[1];
  const float* w_res = (const float*)d_in[2];
  float* out = (float*)d_out;

  // workspace layout (all 16B-aligned section sizes), ~66 MB total
  const size_t nE = (size_t)B_ * HEADS_ * NSEQ * DHEAD;          // 5,386,240
  const size_t nT = (size_t)B_ * HEADS_ * DHEAD * NPAD;          // 5,505,024
  char* p = (char*)d_ws;
  __bf16* qb = (__bf16*)p; p += nE * sizeof(__bf16);
  __bf16* kb = (__bf16*)p; p += nE * sizeof(__bf16);
  __bf16* vt = (__bf16*)p; p += nT * sizeof(__bf16);
  float*  vf = (float*)p;  p += nE * sizeof(float);
  __bf16* xb = (__bf16*)p; p += (size_t)NROWS * DIM * sizeof(__bf16);
  __bf16* wt = (__bf16*)p;

  prep_kernel<<<(NROWS * DIM + 255) / 256, 256, 0, stream>>>(x, w_qkv, xb, wt);

  dim3 g1((NROWS + 15) / 16, NQKV / 256);
  qkv_wmma_kernel<<<g1, 128, 0, stream>>>(xb, wt, qb, kb, vt, vf);

  dim3 g2(QTILES, B_ * HEADS_);
  size_t smem = (size_t)16 * NPAD * (sizeof(float) + sizeof(__bf16)); // 129024
  attn_wmma_kernel<<<g2, 128, smem, stream>>>(qb, kb, vt, out);

  int total4 = (int)(nE / 4);
  conv_add_kernel<<<(total4 + 255) / 256, 256, 0, stream>>>(vf, w_res, out);
}
